// LSTM_80693845557463
// MI455X (gfx1250) — matrix operand order repaired and checked by reading, not compiled, not run on hardware
//
#include <hip/hip_runtime.h>

typedef __attribute__((ext_vector_type(16))) __bf16        v16bf;
typedef __attribute__((ext_vector_type(8)))  float         v8f;
typedef __attribute__((ext_vector_type(8)))  unsigned int  v8u;
typedef __attribute__((ext_vector_type(4)))  float         f4;

#define BSZ 65536
#define HSZ 256
#define BM  128
#define LDS_STRIDE 132   // 128 u32 pairs + 4 pad (bank-conflict avoidance)

__device__ __forceinline__ unsigned int pack_bf16(float a, float b) {
    unsigned int ua = __float_as_uint(a);
    unsigned int ub = __float_as_uint(b);
    ua = (ua + 0x7FFFu + ((ua >> 16) & 1u)) >> 16;   // round-to-nearest-even
    ub = (ub + 0x7FFFu + ((ub >> 16) & 1u)) >> 16;
    return (ua & 0xFFFFu) | (ub << 16);
}

__device__ __forceinline__ float fast_sigmoid(float x) {
    float e = __builtin_amdgcn_exp2f(x * -1.4426950408889634f); // exp(-x)
    return __builtin_amdgcn_rcpf(1.0f + e);
}

__device__ __forceinline__ float fast_tanh(float x) {
    float e = __builtin_amdgcn_exp2f(x * 2.8853900817779268f);  // exp(2x)
    return 1.0f - 2.0f * __builtin_amdgcn_rcpf(e + 1.0f);       // saturates to +/-1
}

// Convert 4 f32 weight matrices [256][256] to packed-bf16-pair layout
// Wp[g][n][kpair]  (u32 = 2 bf16, K ascending) -- matches B-fragment loads.
__global__ __launch_bounds__(256) void prep_weights(
    const float* __restrict__ Wf, const float* __restrict__ Wc,
    const float* __restrict__ Wi, const float* __restrict__ Wo,
    unsigned int* __restrict__ Wp)
{
    int i = blockIdx.x * 256 + threadIdx.x;      // 0 .. 131071
    int g = i >> 15;                             // gate
    int p = i & 32767;                           // pair index = n*128 + kpair
    const float* W = (g == 0) ? Wf : (g == 1) ? Wc : (g == 2) ? Wi : Wo;
    float a = __builtin_nontemporal_load(W + 2 * p);      // f32 weights read once
    float b = __builtin_nontemporal_load(W + 2 * p + 1);
    Wp[i] = pack_bf16(a, b);                              // bf16 image stays cached (RT)
}

__global__ __launch_bounds__(256) void lstm_wmma_kernel(
    const float* __restrict__ input, const float* __restrict__ hidden,
    const float* __restrict__ cell,  const unsigned int* __restrict__ Wp,
    const float* __restrict__ bfv, const float* __restrict__ bcv,
    const float* __restrict__ biv, const float* __restrict__ bov,
    float* __restrict__ out_o, float* __restrict__ out_h, float* __restrict__ out_c)
{
    __shared__ unsigned int lds[BM * LDS_STRIDE];   // combine tile, packed bf16
    __shared__ float sbias[4][HSZ];                 // f, c, i, o biases

    const int tid  = threadIdx.x;
    const int row0 = blockIdx.x * BM;

    // ---- Stage biases once (tid covers 0..255 == HSZ)
    sbias[0][tid] = bfv[tid];
    sbias[1][tid] = bcv[tid];
    sbias[2][tid] = biv[tid];
    sbias[3][tid] = bov[tid];

    // ---- Stage combine = input + hidden as bf16 pairs (row-major, K contiguous).
    // Activations are one-touch streams: non-temporal so weights stay cache-resident.
    for (int i = tid; i < BM * 64; i += 256) {
        const int r = i >> 6;                 // row in tile
        const int q = i & 63;                 // float4 index within row
        const int g = (row0 + r) * HSZ + q * 4;
        f4 a = __builtin_nontemporal_load((const f4*)(input  + g));
        f4 b = __builtin_nontemporal_load((const f4*)(hidden + g));
        f4 s = a + b;
        lds[r * LDS_STRIDE + q * 2 + 0] = pack_bf16(s.x, s.y);
        lds[r * LDS_STRIDE + q * 2 + 1] = pack_bf16(s.z, s.w);
    }
    __syncthreads();

    const int lane = tid & 31;
    const int wv   = tid >> 5;      // wave id 0..7 -> m-tile
    const int nlo  = lane & 15;     // A: row m ; B/C/D: column n (low)
    const int hi   = lane >> 4;     // lane half

    // ---- Load all A fragments (K = 256 -> 8 k-steps) once per wave.
    // A layout: lanes<16 hold K[32s..32s+16); lanes>=16 hold K[32s+16..32s+32).
    v8u afrag[8];
    const unsigned int* arow = &lds[(wv * 16 + nlo) * LDS_STRIDE];
    #pragma unroll
    for (int s = 0; s < 8; ++s) {
        const unsigned int* p = arow + 16 * s + hi * 8;
        uint4 x = *(const uint4*)(p);        // K pairs 0..3  (of this half)
        uint4 y = *(const uint4*)(p + 4);    // K pairs 4..7  (of this half)
        afrag[s] = (v8u){x.x, x.y, x.z, x.w, y.x, y.y, y.z, y.w};
    }

    for (int nt = 0; nt < 16; ++nt) {
        const int n = nt * 16 + nlo;         // output column 0..255

        // B-fragment base rows: B[k][n] = W[n][k] -> read W row n (bf16 pairs).
        const unsigned int* bF = Wp + 0 * 32768 + n * 128;
        const unsigned int* bC = Wp + 1 * 32768 + n * 128;
        const unsigned int* bI = Wp + 2 * 32768 + n * 128;
        const unsigned int* bO = Wp + 3 * 32768 + n * 128;

        v8f accF = {}, accC = {}, accI = {}, accO = {};

        #pragma unroll
        for (int s = 0; s < 8; ++s) {
            v16bf a = __builtin_bit_cast(v16bf, afrag[s]);
            const int off = 16 * s + hi * 8;   // lanes<16: K[32s..+16), lanes>=16: K[+16..+32)

            uint4 x, y; v8u bu; v16bf bb;

            x = *(const uint4*)(bF + off); y = *(const uint4*)(bF + off + 4);
            bu = (v8u){x.x, x.y, x.z, x.w, y.x, y.y, y.z, y.w};
            bb = __builtin_bit_cast(v16bf, bu);
            accF = __builtin_amdgcn_wmma_f32_16x16x32_bf16(false, a, false, bb, (short)0, accF, false, false);

            x = *(const uint4*)(bC + off); y = *(const uint4*)(bC + off + 4);
            bu = (v8u){x.x, x.y, x.z, x.w, y.x, y.y, y.z, y.w};
            bb = __builtin_bit_cast(v16bf, bu);
            accC = __builtin_amdgcn_wmma_f32_16x16x32_bf16(false, a, false, bb, (short)0, accC, false, false);

            x = *(const uint4*)(bI + off); y = *(const uint4*)(bI + off + 4);
            bu = (v8u){x.x, x.y, x.z, x.w, y.x, y.y, y.z, y.w};
            bb = __builtin_bit_cast(v16bf, bu);
            accI = __builtin_amdgcn_wmma_f32_16x16x32_bf16(false, a, false, bb, (short)0, accI, false, false);

            x = *(const uint4*)(bO + off); y = *(const uint4*)(bO + off + 4);
            bu = (v8u){x.x, x.y, x.z, x.w, y.x, y.y, y.z, y.w};
            bb = __builtin_bit_cast(v16bf, bu);
            accO = __builtin_amdgcn_wmma_f32_16x16x32_bf16(false, a, false, bb, (short)0, accO, false, false);
        }

        // ---- Fused gate epilogue. C/D layout: VGPR r -> row (hi*8 + r), lane -> col n.
        const float Bf = sbias[0][n], Bc = sbias[1][n];
        const float Bi = sbias[2][n], Bo = sbias[3][n];
        const int rbase = row0 + wv * 16 + hi * 8;
        #pragma unroll
        for (int r = 0; r < 8; ++r) {
            const int idx = (rbase + r) * HSZ + n;
            float gf = fast_sigmoid(accF[r] + Bf);
            float gc = fast_tanh(accC[r] + Bc);
            float gi = fast_sigmoid(fast_sigmoid(accI[r] + Bi));   // double sigmoid per ref
            float go = fast_sigmoid(accO[r] + Bo);
            float cv = __builtin_nontemporal_load(cell + idx);
            float cs = cv * gf + gc * gi;
            float hs = go * fast_tanh(cs);
            __builtin_nontemporal_store(go, out_o + idx);
            __builtin_nontemporal_store(hs, out_h + idx);
            __builtin_nontemporal_store(cs, out_c + idx);
        }
    }
}

extern "C" void kernel_launch(void* const* d_in, const int* in_sizes, int n_in,
                              void* d_out, int out_size, void* d_ws, size_t ws_size,
                              hipStream_t stream) {
    const float* input  = (const float*)d_in[0];
    const float* hidden = (const float*)d_in[1];
    const float* cell   = (const float*)d_in[2];
    const float* Wf     = (const float*)d_in[3];
    const float* bfv    = (const float*)d_in[4];
    const float* Wc     = (const float*)d_in[5];
    const float* bcv    = (const float*)d_in[6];
    const float* Wi     = (const float*)d_in[7];
    const float* biv    = (const float*)d_in[8];
    const float* Wo     = (const float*)d_in[9];
    const float* bov    = (const float*)d_in[10];

    unsigned int* Wp = (unsigned int*)d_ws;          // 512 KB packed bf16 weights

    float* out_o = (float*)d_out;
    float* out_h = out_o + (size_t)BSZ * HSZ;
    float* out_c = out_h + (size_t)BSZ * HSZ;

    prep_weights<<<512, 256, 0, stream>>>(Wf, Wc, Wi, Wo, Wp);
    lstm_wmma_kernel<<<BSZ / BM, 256, 0, stream>>>(
        input, hidden, cell, Wp, bfv, bcv, biv, bov, out_o, out_h, out_c);
}
